// Model_17343077941634
// MI455X (gfx1250) — matrix operand order repaired and checked by reading, not compiled, not run on hardware
//
#include <hip/hip_runtime.h>
#include <hip/hip_bf16.h>
#include <stdint.h>

typedef __attribute__((ext_vector_type(16))) _Float16     v16h;
typedef __attribute__((ext_vector_type(8)))  float        v8f;
typedef __attribute__((ext_vector_type(4)))  unsigned int u32x4;

union ABReg {
    v16h h;
    u32x4 q[2];
    _Float16 f[16];
};

#define WMMA_F16(A, B, C) \
    __builtin_amdgcn_wmma_f32_16x16x32_f16(false, (A), false, (B), (short)0, (C), false, false)

// ---------------------------------------------------------------------------
// Weight packing (f32 -> zero-padded f16, row-major along K)
// ---------------------------------------------------------------------------
__global__ void pack_w2(const float* __restrict__ w, _Float16* __restrict__ wp) {
    int i = blockIdx.x * 256 + threadIdx.x;            // [64][512]
    if (i >= 64 * 512) return;
    int m = i >> 9, k = i & 511;
    wp[i] = (m < 50 && k < 500) ? (_Float16)w[m * 500 + k] : (_Float16)0.f;
}

__global__ void pack_wp(const float* __restrict__ w, _Float16* __restrict__ wp) {
    int i = blockIdx.x * 256 + threadIdx.x;            // [16][1280]
    if (i >= 16 * 1280) return;
    int m = i / 1280, k = i % 1280;
    wp[i] = (k < 1250) ? (_Float16)w[m * 1250 + k] : (_Float16)0.f;
}

__global__ void pack_fc1(const float* __restrict__ w, _Float16* __restrict__ wp) {
    size_t i = (size_t)blockIdx.x * 256 + threadIdx.x; // [512][3616]
    if (i >= (size_t)512 * 3616) return;
    int n = (int)(i / 3616), k = (int)(i % 3616);
    wp[i] = (n < 500 && k < 3602) ? (_Float16)w[(size_t)n * 3602 + k] : (_Float16)0.f;
}

// ---------------------------------------------------------------------------
// Fused conv1+pool (VALU from LDS) -> conv2 (WMMA, LDS operands) -> pool.
// One workgroup (256 thr = 8 waves) per image.  Dynamic LDS partition:
//   xs   : 2160 f32   =  8,640 B   (input image)
//   h1s  : 8968 f16   = 17,936 B   (pooled conv1 + zero slot @8960)
//   cs   : 14400 f16  = 28,800 B   (conv2 output stage)
//   wlds : 32768 f16  = 65,536 B   (conv2 weights [64][512])
//   total 120,912 B
// ---------------------------------------------------------------------------
#define CB_XS    0
#define CB_H1S   8640
#define CB_CS    26576
#define CB_WLDS  55376
#define CB_LDS   120912

__global__ __launch_bounds__(256) void conv_backbone(
    const float* __restrict__ x,          // [B,1,36,60]
    const float* __restrict__ c1w,        // [20,1,5,5]
    const float* __restrict__ c1b,        // [20]
    const float* __restrict__ c2b,        // [50]
    const _Float16* __restrict__ wpad2,   // [64][512]
    _Float16* __restrict__ feat)          // [B][3616]
{
    extern __shared__ char smem[];
    float*    xs   = (float*)(smem + CB_XS);
    _Float16* h1s  = (_Float16*)(smem + CB_H1S);
    _Float16* cs   = (_Float16*)(smem + CB_CS);
    _Float16* wlds = (_Float16*)(smem + CB_WLDS);

    const int b = blockIdx.x;
    const int tid = threadIdx.x;

    // stage input image + conv2 weights in LDS
    const float* xb = x + (size_t)b * 36 * 60;
    for (int i = tid; i < 2160; i += 256) xs[i] = xb[i];
    {
        const u32x4* src = (const u32x4*)wpad2;
        u32x4* dst = (u32x4*)wlds;
        for (int i = tid; i < 4096; i += 256) dst[i] = src[i];
    }
    if (tid == 0) h1s[8960] = (_Float16)0.f;   // zero slot for padded K gathers
    __syncthreads();

    // conv1 (valid 5x5, 1->20 ch) + 2x2 maxpool, into LDS as f16
    for (int o = tid; o < 20 * 16 * 28; o += 256) {
        int oc = o / 448, rem = o % 448, py = rem / 28, px = rem % 28;
        const float* w = c1w + oc * 25;
        float m = -3.4e38f;
        for (int dy = 0; dy < 2; ++dy)
            for (int dx = 0; dx < 2; ++dx) {
                int iy0 = 2 * py + dy, ix0 = 2 * px + dx;
                float s = 0.f;
#pragma unroll
                for (int ky = 0; ky < 5; ++ky)
#pragma unroll
                    for (int kx = 0; kx < 5; ++kx)
                        s += xs[(iy0 + ky) * 60 + ix0 + kx] * w[ky * 5 + kx];
                m = fmaxf(m, s);
            }
        h1s[o] = (_Float16)(m + c1b[oc]);
    }
    __syncthreads();

    // conv2 as GEMM: M=64(pad of 50) x K=512(pad of 500) x N=288 pixels
    const int wave = tid >> 5, lane = tid & 31;
    const int n = lane & 15, g = lane >> 4;

    for (int nt = wave; nt < 18; nt += 8) {
        const int p = nt * 16 + n;             // output pixel 0..287
        const int py = p / 24, px = p % 24;
        const int pbase = py * 28 + px;

        v8f accv[4] = {{0}, {0}, {0}, {0}};
        for (int ks = 0; ks < 16; ++ks) {
            const int kb2 = ks * 32 + 8 * g;
            // B tile: branch-free per-lane gather (16x ds_load_u16)
            ABReg bb;
#pragma unroll
            for (int i = 0; i < 16; ++i) {
                int k = kb2 + i + (i >= 8 ? 8 : 0);
                int ic = k / 25, r = k - ic * 25;
                int ky = r / 5,  kx = r - ky * 5;
                int off = ic * 448 + ky * 28 + kx + pbase;
                int addr = (k < 500) ? off : 8960;
                bb.f[i] = h1s[addr];
            }
            // A tiles from LDS weights: 2x ds_load_b128 each
            const int kbase = ks * 32;
            ABReg av0, av1, av2, av3;
            {
                const _Float16* ap = wlds + (size_t)(0 * 16 + n) * 512 + kbase;
                av0.q[0] = *(const u32x4*)(ap + 8 * g);
                av0.q[1] = *(const u32x4*)(ap + 16 + 8 * g);
            }
            {
                const _Float16* ap = wlds + (size_t)(1 * 16 + n) * 512 + kbase;
                av1.q[0] = *(const u32x4*)(ap + 8 * g);
                av1.q[1] = *(const u32x4*)(ap + 16 + 8 * g);
            }
            {
                const _Float16* ap = wlds + (size_t)(2 * 16 + n) * 512 + kbase;
                av2.q[0] = *(const u32x4*)(ap + 8 * g);
                av2.q[1] = *(const u32x4*)(ap + 16 + 8 * g);
            }
            {
                const _Float16* ap = wlds + (size_t)(3 * 16 + n) * 512 + kbase;
                av3.q[0] = *(const u32x4*)(ap + 8 * g);
                av3.q[1] = *(const u32x4*)(ap + 16 + 8 * g);
            }
            accv[0] = WMMA_F16(av0.h, bb.h, accv[0]);
            accv[1] = WMMA_F16(av1.h, bb.h, accv[1]);
            accv[2] = WMMA_F16(av2.h, bb.h, accv[2]);
            accv[3] = WMMA_F16(av3.h, bb.h, accv[3]);
        }
        // C layout: VGPR r -> M = r + 8*g, N = lane&15
#pragma unroll
        for (int mt = 0; mt < 4; ++mt) {
#pragma unroll
            for (int r = 0; r < 8; ++r) {
                int oc = mt * 16 + r + 8 * g;
                if (oc < 50) cs[oc * 288 + p] = (_Float16)(accv[mt][r] + c2b[oc]);
            }
        }
    }
    __syncthreads();

    // 2x2 maxpool 12x24 -> 6x12, write f16 features [oc*72 + y*12 + x]
    _Float16* fb = feat + (size_t)b * 3616;
    for (int o = tid; o < 3600; o += 256) {
        int oc = o / 72, rem = o % 72, yy = rem / 12, xx = rem % 12;
        int base = oc * 288 + (2 * yy) * 24 + 2 * xx;
        float v00 = (float)cs[base], v01 = (float)cs[base + 1];
        float v10 = (float)cs[base + 24], v11 = (float)cs[base + 25];
        fb[o] = (_Float16)fmaxf(fmaxf(v00, v01), fmaxf(v10, v11));
    }
}

// ---------------------------------------------------------------------------
// Primary caps conv: 50->16 ch, 5x5, pad 2, on 6x12. WMMA M=16, K=1280, N=80.
// Activations + weights staged in LDS; branch-free gathers via zero slot.
// ---------------------------------------------------------------------------
__global__ __launch_bounds__(256) void pcaps_conv(
    const _Float16* __restrict__ feat,   // h2 in f16 = feat[b][oc*72+px]
    const _Float16* __restrict__ wpadP,  // [16][1280]
    const float* __restrict__ pb,        // [16]
    float* __restrict__ pbuf)            // [B][16][72]
{
    __shared__ _Float16 h2s[3616];       // 3600 + zero slot @3600
    __shared__ _Float16 wls[16 * 1280];  // 40,960 B
    const int b = blockIdx.x, tid = threadIdx.x;

    {
        const u32x4* src = (const u32x4*)(feat + (size_t)b * 3616);
        u32x4* dst = (u32x4*)h2s;
        for (int i = tid; i < 450; i += 256) dst[i] = src[i];   // 3600 f16
        const u32x4* wsrc = (const u32x4*)wpadP;
        u32x4* wdst = (u32x4*)wls;
        for (int i = tid; i < 2560; i += 256) wdst[i] = wsrc[i];
        if (tid < 16) h2s[3600 + tid] = (_Float16)0.f;
    }
    __syncthreads();

    const int wave = tid >> 5, lane = tid & 31;
    if (wave >= 5) return;                 // wave-uniform: EXEC stays full
    const int n = lane & 15, g = lane >> 4;
    const int p = wave * 16 + n;           // 0..79 (valid < 72)
    const int pc = (p < 72) ? p : 71;      // clamp so all lanes execute WMMA
    const int py = pc / 12, px = pc % 12;

    v8f acc = {0};
    for (int ks = 0; ks < 40; ++ks) {
        const int kbase = ks * 32;
        const int kb2 = kbase + 8 * g;
        ABReg bb, av;
#pragma unroll
        for (int i = 0; i < 16; ++i) {
            int k = kb2 + i + (i >= 8 ? 8 : 0);
            int ic = k / 25, r = k - ic * 25;
            int ky = r / 5,  kx = r - ky * 5;
            int yy = py + ky - 2, xx = px + kx - 2;
            bool ok = (k < 1250) & ((unsigned)yy < 6u) & ((unsigned)xx < 12u);
            int addr = ok ? (ic * 72 + yy * 12 + xx) : 3600;
            bb.f[i] = h2s[addr];
        }
        const _Float16* ap = wls + (size_t)n * 1280 + kbase;
        av.q[0] = *(const u32x4*)(ap + 8 * g);
        av.q[1] = *(const u32x4*)(ap + 16 + 8 * g);
        acc = WMMA_F16(av.h, bb.h, acc);
    }
    if (p < 72) {
        float* pob = pbuf + (size_t)b * 1152;
#pragma unroll
        for (int r = 0; r < 8; ++r) {
            int oc = r + 8 * g;
            pob[oc * 72 + p] = acc[r] + pb[oc];
        }
    }
}

// ---------------------------------------------------------------------------
// Squash + priors + 3 dynamic-routing iterations (LDS atomics for reductions)
// ---------------------------------------------------------------------------
__global__ __launch_bounds__(256) void routing(
    const float* __restrict__ pbuf,   // [B][16][72]
    const float* __restrict__ rw,     // [2,288,4,1]
    _Float16* __restrict__ feat)      // writes feat[b][3600..3615]
{
    __shared__ float pri[2 * 288];
    __shared__ float lg[2 * 288];
    __shared__ float acc[2], sacc[2], vsh[2];
    const int b = blockIdx.x, tid = threadIdx.x;
    const float* pob = pbuf + (size_t)b * 1152;

    for (int r = tid; r < 288; r += 256) {
        int blk = r / 72, px = r % 72;
        float u0 = pob[(0 * 4 + blk) * 72 + px];
        float u1 = pob[(1 * 4 + blk) * 72 + px];
        float u2 = pob[(2 * 4 + blk) * 72 + px];
        float u3 = pob[(3 * 4 + blk) * 72 + px];
        float n2 = u0 * u0 + u1 * u1 + u2 * u2 + u3 * u3;
        float sc = (n2 / (1.f + n2)) * rsqrtf(n2);
        u0 *= sc; u1 *= sc; u2 *= sc; u3 *= sc;
#pragma unroll
        for (int k = 0; k < 2; ++k) {
            const float* w = rw + ((size_t)k * 288 + r) * 4;
            pri[k * 288 + r] = u0 * w[0] + u1 * w[1] + u2 * w[2] + u3 * w[3];
            lg[k * 288 + r] = 0.f;
        }
    }
    __syncthreads();

    for (int it = 0; it < 3; ++it) {
        if (tid < 2) { acc[tid] = 0.f; sacc[tid] = 0.f; }
        __syncthreads();
        for (int r = tid; r < 288; r += 256)
            for (int k = 0; k < 2; ++k)
                atomicAdd(&acc[k], __expf(lg[k * 288 + r]));
        __syncthreads();
        for (int r = tid; r < 288; r += 256)
            for (int k = 0; k < 2; ++k)
                atomicAdd(&sacc[k], (__expf(lg[k * 288 + r]) / acc[k]) * pri[k * 288 + r]);
        __syncthreads();
        if (tid < 2) {
            float s = sacc[tid];
            vsh[tid] = s * fabsf(s) / (1.f + s * s);   // squash of a scalar
        }
        __syncthreads();
        if (it < 2) {
            for (int r = tid; r < 288; r += 256)
                for (int k = 0; k < 2; ++k)
                    lg[k * 288 + r] += pri[k * 288 + r] * vsh[k];
            __syncthreads();
        }
    }
    if (tid < 2)  feat[(size_t)b * 3616 + 3600 + tid] = (_Float16)vsh[tid];
    if (tid >= 2 && tid < 16) feat[(size_t)b * 3616 + 3600 + tid] = (_Float16)0.f;
}

// ---------------------------------------------------------------------------
// FC1: 2048 x 512(pad of 500) x 3616(pad of 3602), WMMA, double-buffered
// global b128 operand loads, bias + ReLU.
// ---------------------------------------------------------------------------
__global__ __launch_bounds__(256) void fc1_gemm(
    const _Float16* __restrict__ feat,   // [2048][3616]
    const _Float16* __restrict__ fc1wp,  // [512][3616]
    const float* __restrict__ fc1b,      // [500]
    float* __restrict__ f1)              // [2048][512]
{
    const int tid = threadIdx.x, wave = tid >> 5, lane = tid & 31;
    const int tile = blockIdx.x * 8 + wave;     // 0..4095
    const int mtile = tile >> 5, ntile = tile & 31;
    const int n = lane & 15, g = lane >> 4;
    const _Float16* arow = feat + (size_t)(mtile * 16 + n) * 3616;
    const _Float16* brow = fc1wp + (size_t)(ntile * 16 + n) * 3616;

#define LOADT(AV, BV, KB) do {                                   \
        (AV).q[0] = *(const u32x4*)(arow + (KB) + 8 * g);        \
        (AV).q[1] = *(const u32x4*)(arow + (KB) + 16 + 8 * g);   \
        (BV).q[0] = *(const u32x4*)(brow + (KB) + 8 * g);        \
        (BV).q[1] = *(const u32x4*)(brow + (KB) + 16 + 8 * g);   \
    } while (0)

    v8f acc = {0};
    ABReg a0, b0, a1, b1;
    LOADT(a0, b0, 0);
    for (int ks = 0; ks < 112; ks += 2) {        // 113 k-steps total
        LOADT(a1, b1, (ks + 1) * 32);
        acc = WMMA_F16(a0.h, b0.h, acc);
        LOADT(a0, b0, (ks + 2) * 32);            // ks+2 <= 112 always valid
        acc = WMMA_F16(a1.h, b1.h, acc);
    }
    acc = WMMA_F16(a0.h, b0.h, acc);             // ks = 112
#undef LOADT

    const int col = ntile * 16 + n;
    const float bias = (col < 500) ? fc1b[col] : 0.f;
#pragma unroll
    for (int r = 0; r < 8; ++r) {
        int row = mtile * 16 + r + 8 * g;
        f1[(size_t)row * 512 + col] = fmaxf(acc[r] + bias, 0.f);
    }
}

// ---------------------------------------------------------------------------
// FC2: out[b][o] = sum_j f1[b][j]*w[o][j] + w[o][500..501].y[b] + b[o]
// ---------------------------------------------------------------------------
__global__ __launch_bounds__(256) void fc2_kernel(
    const float* __restrict__ f1, const float* __restrict__ y,
    const float* __restrict__ w2, const float* __restrict__ b2,
    float* __restrict__ out)
{
    int id = blockIdx.x * 256 + threadIdx.x;
    if (id >= 2048 * 2) return;
    int b = id >> 1, o = id & 1;
    const float* wr = w2 + o * 502;
    const float* fr = f1 + (size_t)b * 512;
    float s = b2[o];
    for (int j = 0; j < 500; ++j) s += fr[j] * wr[j];
    s += wr[500] * y[b * 2 + 0] + wr[501] * y[b * 2 + 1];
    out[id] = s;
}

// ---------------------------------------------------------------------------
extern "C" void kernel_launch(void* const* d_in, const int* in_sizes, int n_in,
                              void* d_out, int out_size, void* d_ws, size_t ws_size,
                              hipStream_t stream)
{
    const float* x   = (const float*)d_in[0];
    const float* y   = (const float*)d_in[1];
    const float* c1w = (const float*)d_in[2];
    const float* c1b = (const float*)d_in[3];
    const float* c2w = (const float*)d_in[4];
    const float* c2b = (const float*)d_in[5];
    const float* pw  = (const float*)d_in[6];
    const float* pb  = (const float*)d_in[7];
    const float* rw  = (const float*)d_in[8];
    const float* f1w = (const float*)d_in[9];
    const float* f1b = (const float*)d_in[10];
    const float* f2w = (const float*)d_in[11];
    const float* f2b = (const float*)d_in[12];
    float* out = (float*)d_out;

    char* ws = (char*)d_ws;
    _Float16* wpad2 = (_Float16*)(ws + 0);          //    65,536 B
    _Float16* wpadP = (_Float16*)(ws + 65536);      //    40,960 B
    _Float16* fc1wp = (_Float16*)(ws + 106496);     // 3,702,784 B
    _Float16* feat  = (_Float16*)(ws + 3809280);    // 14,811,136 B
    float*    pbuf  = (float*)   (ws + 18620416);   // 9,437,184 B
    float*    f1    = (float*)   (ws + 28057600);   // 4,194,304 B (end ~32.3 MB)

    pack_w2 <<<(64 * 512 + 255) / 256, 256, 0, stream>>>(c2w, wpad2);
    pack_wp <<<(16 * 1280 + 255) / 256, 256, 0, stream>>>(pw, wpadP);
    pack_fc1<<<(int)(((size_t)512 * 3616 + 255) / 256), 256, 0, stream>>>(f1w, fc1wp);

    conv_backbone<<<2048, 256, CB_LDS, stream>>>(x, c1w, c1b, c2b, wpad2, feat);
    pcaps_conv  <<<2048, 256, 0, stream>>>(feat, wpadP, pb, pbuf);
    routing     <<<2048, 256, 0, stream>>>(pbuf, rw, feat);
    fc1_gemm    <<<512, 256, 0, stream>>>(feat, fc1wp, f1b, f1);
    fc2_kernel  <<<16, 256, 0, stream>>>(f1, y, f2w, f2b, out);

    (void)in_sizes; (void)n_in; (void)out_size; (void)ws_size;
}
